// DIS_43937515438592
// MI455X (gfx1250) — matrix operand order repaired and checked by reading, not compiled, not run on hardware
//
#include <hip/hip_runtime.h>
#include <cstdint>
#include <cstddef>

#define NN     100000
#define KK     8
#define EE     100000
#define NFEATD 500
#define NHIDD  128
#define NCLS   10

typedef __attribute__((ext_vector_type(16))) _Float16 v16h;
typedef __attribute__((ext_vector_type(8)))  float    v8f;

union HFrag { v16h h; uint4 u[2]; };

// ---------------------------------------------------------------------------
// CDNA5 packed-f16 global atomic add (2 halves per op) — halves atomic count
// vs per-float atomics on the scatter path. Device scope for cross-WGP vis.
// ---------------------------------------------------------------------------
__device__ __forceinline__ void atomic_pk_add_f16(unsigned int* addr, unsigned int data) {
  asm volatile("global_atomic_pk_add_f16 %0, %1, off scope:SCOPE_DEV"
               :
               : "v"((unsigned long long)(uintptr_t)addr), "v"(data)
               : "memory");
}

// ---------------------------------------------------------------------------
// WMMA fragment helpers (layouts per CDNA5 ISA §7.12.2, wave32)
// A 16x32 f16: lane row = lane&15; lane holds K runs [ld*8..+7] and [16+ld*8..+7]
// B 32x16 f16: lane col = lane&15; lane holds K runs [ (lane>>4)*8 .. +7 ] and [ 16+(lane>>4)*8 .. +7 ]
// C/D 16x16 f32: VGPR r -> row (lane>>4)*8 + r, col = lane&15
// ---------------------------------------------------------------------------
__device__ __forceinline__ v16h load_a_f16(const _Float16* __restrict__ A, int row,
                                           int lda, int kBase, int lane) {
  int ld = lane >> 4;
  const _Float16* rp = A + (size_t)row * lda + kBase + ld * 8;
  HFrag F;
  F.u[0] = *(const uint4*)(rp);
  F.u[1] = *(const uint4*)(rp + 16);
  return F.h;
}

__device__ __forceinline__ v16h load_a_f32(const float* __restrict__ A, int row,
                                           int lda, int kBase, int klim, int lane) {
  int ld = lane >> 4;
  const float* rp = A + (size_t)row * lda;
  int k0 = kBase + ld * 8;
  int k1 = k0 + 16;
  float t[16];
  if (kBase + 32 <= klim) {
    float4 a0 = *(const float4*)(rp + k0);
    float4 a1 = *(const float4*)(rp + k0 + 4);
    float4 a2 = *(const float4*)(rp + k1);
    float4 a3 = *(const float4*)(rp + k1 + 4);
    t[0]=a0.x; t[1]=a0.y; t[2]=a0.z;  t[3]=a0.w;
    t[4]=a1.x; t[5]=a1.y; t[6]=a1.z;  t[7]=a1.w;
    t[8]=a2.x; t[9]=a2.y; t[10]=a2.z; t[11]=a2.w;
    t[12]=a3.x;t[13]=a3.y;t[14]=a3.z; t[15]=a3.w;
  } else {
#pragma unroll
    for (int j = 0; j < 8; ++j) t[j]     = (k0 + j < klim) ? rp[k0 + j] : 0.0f;
#pragma unroll
    for (int j = 0; j < 8; ++j) t[8 + j] = (k1 + j < klim) ? rp[k1 + j] : 0.0f;
  }
  v16h F;
#pragma unroll
  for (int j = 0; j < 16; ++j) F[j] = (_Float16)t[j];
  return F;
}

// one K-tile against all 8 column tiles (16x128 strip of D)
__device__ __forceinline__ void wmma_row8(const v16h& a, const uint4* __restrict__ bbase,
                                          int lane, v8f acc[8]) {
#pragma unroll
  for (int nt = 0; nt < 8; ++nt) {
    HFrag B;
    const uint4* p = bbase + nt * 64 + lane * 2;   // frag = 32 lanes * 32B
    B.u[0] = p[0];
    B.u[1] = p[1];
    acc[nt] = __builtin_amdgcn_wmma_f32_16x16x32_f16(false, a, false, B.h,
                                                     (short)0, acc[nt], false, false);
  }
}

__device__ __forceinline__ void store_tile(const v8f acc[8], _Float16* __restrict__ D,
                                           int mBase, int lane,
                                           const float* __restrict__ bias, bool do_relu) {
  int ld  = lane >> 4;
  int col = lane & 15;
#pragma unroll
  for (int nt = 0; nt < 8; ++nt) {
    float b = bias[nt * 16 + col];
#pragma unroll
    for (int r = 0; r < 8; ++r) {
      float v = acc[nt][r] + b;
      if (do_relu) v = fmaxf(v, 0.0f);
      D[(size_t)(mBase + ld * 8 + r) * NHIDD + nt * 16 + col] = (_Float16)v;
    }
  }
}

// ---------------------------------------------------------------------------
// Weight repack: fp32 row-major [Ksrc x 128] -> B-fragment-ordered f16,
// K zero-padded to Kpad (total = Kpad*128 elements).
// packed idx i = ((kt*8+nt)*32 + lane)*16 + j  ->  W[kt*32+(j>>3)*16+(lane>>4)*8+(j&7)][nt*16+(lane&15)]
// ---------------------------------------------------------------------------
__global__ void k_pack(const float* __restrict__ W, _Float16* __restrict__ Wp,
                       int Ksrc, int total) {
  int i = blockIdx.x * blockDim.x + threadIdx.x;
  if (i >= total) return;
  int j    = i & 15;
  int lane = (i >> 4) & 31;
  int frag = i >> 9;
  int nt = frag & 7, kt = frag >> 3;
  int kk = kt * 32 + (j >> 3) * 16 + (lane >> 4) * 8 + (j & 7);
  int n  = nt * 16 + (lane & 15);
  float v = (kk < Ksrc) ? W[(size_t)kk * NHIDD + n] : 0.0f;
  Wp[i] = (_Float16)v;
}

// ---------------------------------------------------------------------------
// Edge-count and reciprocal (shared by both layers)
// ---------------------------------------------------------------------------
__global__ void k_count(const int* __restrict__ edges, float* __restrict__ cnt) {
  int i = blockIdx.x * blockDim.x + threadIdx.x;
  if (i >= KK * EE) return;
  int k = i / EE, e = i - k * EE;
  int dst = edges[(size_t)k * 2 * EE + EE + e];
  atomicAdd(cnt + (size_t)k * NN + dst, 1.0f);
}

__global__ void k_inv(float* __restrict__ c) {
  int i = blockIdx.x * blockDim.x + threadIdx.x;
  if (i >= KK * NN) return;
  c[i] = 1.0f / fmaxf(c[i], 1.0f);
}

// ---------------------------------------------------------------------------
// Scatter: one wave per edge; 128 f16 = 64 dwords -> 2 pk-atomics per lane
// ---------------------------------------------------------------------------
__global__ void __launch_bounds__(256) k_scatter(const int* __restrict__ edges,
                                                 const _Float16* __restrict__ H,
                                                 _Float16* __restrict__ Agg) {
  unsigned int wave = (blockIdx.x * 256u + threadIdx.x) >> 5;
  int lane = threadIdx.x & 31;
  int k = wave / EE;
  int e = wave - k * EE;
  int src = edges[(size_t)k * 2 * EE + e];
  int dst = edges[(size_t)k * 2 * EE + EE + e];
  const unsigned int* hr = (const unsigned int*)(H + (size_t)src * NHIDD);
  unsigned int* ar = (unsigned int*)(Agg + ((size_t)k * NN + dst) * NHIDD);
  unsigned int d0 = hr[lane];
  unsigned int d1 = hr[lane + 32];
  atomic_pk_add_f16(ar + lane, d0);
  atomic_pk_add_f16(ar + lane + 32, d1);
}

// ---------------------------------------------------------------------------
// GEMM 1: h0 = x @ W_in + b_in   (A fp32, K=500 handled with tail tile)
// ---------------------------------------------------------------------------
__global__ void __launch_bounds__(256) k_gemm_in(const float* __restrict__ X,
                                                 const _Float16* __restrict__ Wp,
                                                 const float* __restrict__ bias,
                                                 _Float16* __restrict__ D) {
  int lane  = threadIdx.x & 31;
  int mBase = blockIdx.x * 128 + (threadIdx.x >> 5) * 16;
  if (mBase >= NN) return;
  int row = mBase + (lane & 15);
  v8f zero = {};
  v8f acc[8];
#pragma unroll
  for (int nt = 0; nt < 8; ++nt) acc[nt] = zero;
  for (int kt = 0; kt < 16; ++kt) {     // Kpad = 512
    v16h a = load_a_f32(X, row, NFEATD, kt * 32, NFEATD, lane);
    wmma_row8(a, (const uint4*)Wp + (size_t)kt * 512, lane, acc);
  }
  store_tile(acc, D, mBase, lane, bias, false);
}

// ---------------------------------------------------------------------------
// Conv GEMM: h' = relu([h | mean_agg_0..7] @ Wc + bc)  — concat never built;
// K=1152 streamed from 9 blocks, agg blocks scaled by 1/cnt per row.
// ---------------------------------------------------------------------------
__global__ void __launch_bounds__(256) k_gemm_conv(const _Float16* __restrict__ Hin,
                                                   const _Float16* __restrict__ Agg,
                                                   const float* __restrict__ Inv,
                                                   const _Float16* __restrict__ Wp,
                                                   const float* __restrict__ bias,
                                                   _Float16* __restrict__ D) {
  int lane  = threadIdx.x & 31;
  int mBase = blockIdx.x * 128 + (threadIdx.x >> 5) * 16;
  if (mBase >= NN) return;
  int row = mBase + (lane & 15);
  v8f zero = {};
  v8f acc[8];
#pragma unroll
  for (int nt = 0; nt < 8; ++nt) acc[nt] = zero;
  for (int kt = 0; kt < 36; ++kt) {
    int blk  = kt >> 2;
    int koff = (kt & 3) * 32;
    v16h a;
    if (blk == 0) {
      a = load_a_f16(Hin, row, NHIDD, koff, lane);
    } else {
      a = load_a_f16(Agg + (size_t)(blk - 1) * NN * NHIDD, row, NHIDD, koff, lane);
      _Float16 s = (_Float16)Inv[(size_t)(blk - 1) * NN + row];
#pragma unroll
      for (int j = 0; j < 16; ++j) a[j] = (_Float16)(a[j] * s);
    }
    wmma_row8(a, (const uint4*)Wp + (size_t)kt * 512, lane, acc);
  }
  store_tile(acc, D, mBase, lane, bias, true);
}

// ---------------------------------------------------------------------------
// Classifier GEMM: z = relu([h0|h1|h2] @ W_cls1 + b1), K=384
// ---------------------------------------------------------------------------
__global__ void __launch_bounds__(256) k_gemm_cls(const _Float16* __restrict__ H0,
                                                  const _Float16* __restrict__ H1,
                                                  const _Float16* __restrict__ H2,
                                                  const _Float16* __restrict__ Wp,
                                                  const float* __restrict__ bias,
                                                  _Float16* __restrict__ D) {
  int lane  = threadIdx.x & 31;
  int mBase = blockIdx.x * 128 + (threadIdx.x >> 5) * 16;
  if (mBase >= NN) return;
  int row = mBase + (lane & 15);
  const _Float16* srcs[3] = {H0, H1, H2};
  v8f zero = {};
  v8f acc[8];
#pragma unroll
  for (int nt = 0; nt < 8; ++nt) acc[nt] = zero;
  for (int kt = 0; kt < 12; ++kt) {
    v16h a = load_a_f16(srcs[kt >> 2], row, NHIDD, (kt & 3) * 32, lane);
    wmma_row8(a, (const uint4*)Wp + (size_t)kt * 512, lane, acc);
  }
  store_tile(acc, D, mBase, lane, bias, true);
}

// ---------------------------------------------------------------------------
// Head: y = z @ W_cls2 + b2 ; log_softmax.  W (128x10) staged in LDS.
// ---------------------------------------------------------------------------
__global__ void __launch_bounds__(256) k_final(const _Float16* __restrict__ Z,
                                               const float* __restrict__ W2,
                                               const float* __restrict__ b2,
                                               float* __restrict__ out) {
  __shared__ float sW[NHIDD * NCLS];
  for (int i = threadIdx.x; i < NHIDD * NCLS; i += 256) sW[i] = W2[i];
  __syncthreads();
  int row = blockIdx.x * 256 + threadIdx.x;
  if (row >= NN) return;
  float y[NCLS];
#pragma unroll
  for (int c = 0; c < NCLS; ++c) y[c] = b2[c];
  const uint4* zr = (const uint4*)(Z + (size_t)row * NHIDD);
  for (int q4 = 0; q4 < 16; ++q4) {
    uint4 u = zr[q4];
    unsigned int w[4] = {u.x, u.y, u.z, u.w};
#pragma unroll
    for (int d = 0; d < 4; ++d) {
      union { unsigned int u32; _Float16 h[2]; } cv;
      cv.u32 = w[d];
      float z0 = (float)cv.h[0], z1 = (float)cv.h[1];
      int q = q4 * 8 + d * 2;
#pragma unroll
      for (int c = 0; c < NCLS; ++c)
        y[c] += z0 * sW[q * NCLS + c] + z1 * sW[(q + 1) * NCLS + c];
    }
  }
  float m = y[0];
#pragma unroll
  for (int c = 1; c < NCLS; ++c) m = fmaxf(m, y[c]);
  float s = 0.0f;
#pragma unroll
  for (int c = 0; c < NCLS; ++c) s += expf(y[c] - m);
  float l = logf(s);
#pragma unroll
  for (int c = 0; c < NCLS; ++c) out[(size_t)row * NCLS + c] = y[c] - m - l;
}

// ---------------------------------------------------------------------------
extern "C" void kernel_launch(void* const* d_in, const int* in_sizes, int n_in,
                              void* d_out, int out_size, void* d_ws, size_t ws_size,
                              hipStream_t stream) {
  (void)in_sizes; (void)n_in; (void)out_size; (void)ws_size;
  const float* x      = (const float*)d_in[0];
  const int*   edges  = (const int*)  d_in[1];
  const float* W_in   = (const float*)d_in[2];
  const float* b_in   = (const float*)d_in[3];
  const float* W_conv = (const float*)d_in[4];
  const float* b_conv = (const float*)d_in[5];
  const float* W_cls1 = (const float*)d_in[6];
  const float* b_cls1 = (const float*)d_in[7];
  const float* W_cls2 = (const float*)d_in[8];
  const float* b_cls2 = (const float*)d_in[9];
  float* out = (float*)d_out;

  char* ws = (char*)d_ws;
  size_t off = 0;
  auto carve = [&](size_t bytes) {
    char* p = ws + off;
    off += (bytes + 255) & ~(size_t)255;
    return p;
  };
  _Float16* h0   = (_Float16*)carve((size_t)NN * NHIDD * 2);
  _Float16* h1   = (_Float16*)carve((size_t)NN * NHIDD * 2);
  _Float16* h2   = (_Float16*)carve((size_t)NN * NHIDD * 2);
  _Float16* z    = (_Float16*)carve((size_t)NN * NHIDD * 2);
  _Float16* agg  = (_Float16*)carve((size_t)KK * NN * NHIDD * 2);
  float*    inv  = (float*)   carve((size_t)KK * NN * 4);
  _Float16* WpIn = (_Float16*)carve((size_t)512  * NHIDD * 2);
  _Float16* WpC0 = (_Float16*)carve((size_t)1152 * NHIDD * 2);
  _Float16* WpC1 = (_Float16*)carve((size_t)1152 * NHIDD * 2);
  _Float16* WpL1 = (_Float16*)carve((size_t)384  * NHIDD * 2);

  const int TB = 256;
  const dim3 gemmGrid((NN + 127) / 128);

  // weight repack (tiny, once per call)
  k_pack<<<dim3((512  * NHIDD + TB - 1) / TB), dim3(TB), 0, stream>>>(W_in,   WpIn, NFEATD, 512  * NHIDD);
  k_pack<<<dim3((1152 * NHIDD + TB - 1) / TB), dim3(TB), 0, stream>>>(W_conv, WpC0, 1152,   1152 * NHIDD);
  k_pack<<<dim3((1152 * NHIDD + TB - 1) / TB), dim3(TB), 0, stream>>>(W_conv + (size_t)1152 * NHIDD,
                                                                      WpC1, 1152, 1152 * NHIDD);
  k_pack<<<dim3((384  * NHIDD + TB - 1) / TB), dim3(TB), 0, stream>>>(W_cls1, WpL1, 384, 384 * NHIDD);

  // per-(relation,node) edge counts -> reciprocal means (layer-invariant)
  hipMemsetAsync(inv, 0, (size_t)KK * NN * 4, stream);
  k_count<<<dim3((KK * EE + TB - 1) / TB), dim3(TB), 0, stream>>>(edges, inv);
  k_inv  <<<dim3((KK * NN + TB - 1) / TB), dim3(TB), 0, stream>>>(inv);

  // input projection
  k_gemm_in<<<gemmGrid, dim3(TB), 0, stream>>>(x, WpIn, b_in, h0);

  // layer 0
  hipMemsetAsync(agg, 0, (size_t)KK * NN * NHIDD * 2, stream);
  k_scatter<<<dim3(KK * EE * 32 / TB), dim3(TB), 0, stream>>>(edges, h0, agg);
  k_gemm_conv<<<gemmGrid, dim3(TB), 0, stream>>>(h0, agg, inv, WpC0, b_conv, h1);

  // layer 1
  hipMemsetAsync(agg, 0, (size_t)KK * NN * NHIDD * 2, stream);
  k_scatter<<<dim3(KK * EE * 32 / TB), dim3(TB), 0, stream>>>(edges, h1, agg);
  k_gemm_conv<<<gemmGrid, dim3(TB), 0, stream>>>(h1, agg, inv, WpC1, b_conv + NHIDD, h2);

  // JK classifier + head
  k_gemm_cls<<<gemmGrid, dim3(TB), 0, stream>>>(h0, h1, h2, WpL1, b_cls1, z);
  k_final<<<dim3((NN + TB - 1) / TB), dim3(TB), 0, stream>>>(z, W_cls2, b_cls2, out);
}
